// Prediction_86053964742793
// MI455X (gfx1250) — matrix operand order repaired and checked by reading, not compiled, not run on hardware
//
#include <hip/hip_runtime.h>
#include <hip/hip_bf16.h>

// Problem constants (match reference)
#define B_ 256
#define S_ 256
#define I_ 64
#define H_ 512
#define P_ 32

typedef _Float16 half_t;
typedef __attribute__((ext_vector_type(16))) _Float16 v16h;
typedef __attribute__((ext_vector_type(8)))  float    v8f;

union Frag16 { v16h v; uint4 u[2]; };

// ---------------------------------------------------------------------------
// f32 -> f16 conversion (weights / input, done once per launch)
// ---------------------------------------------------------------------------
__global__ void f32_to_f16_kernel(const float* __restrict__ src,
                                  half_t* __restrict__ dst, int n) {
    int i = blockIdx.x * blockDim.x + threadIdx.x;
    if (i < n) dst[i] = (half_t)src[i];
}

// ---------------------------------------------------------------------------
// Register-blocked WMMA K-loop: one wave accumulates a 32x32 output block as
// a 2x2 grid of 16x16 tiles. Each A fragment is reused against 2 B fragments
// and vice versa -> 4 v_wmma per 8 b128 loads (2x less L2 traffic than 1x1).
// a*/b* pointers already include the per-lane (sel) shifts.
// ---------------------------------------------------------------------------
__device__ inline void wmma2x2_k_loop(const half_t* __restrict__ a0p,
                                      const half_t* __restrict__ a1p,
                                      const half_t* __restrict__ b0p,
                                      const half_t* __restrict__ b1p,
                                      int K, v8f acc[2][2]) {
    for (int k0 = 0; k0 < K; k0 += 32) {
        Frag16 a0, a1, b0, b1;
        // A 16x32 f16 fragment (ISA 7.12.2): lane<16 -> K {0..7,16..23},
        // lane>=16 -> +8 (folded into a*p).
        a0.u[0] = *(const uint4*)(a0p + k0);
        a0.u[1] = *(const uint4*)(a0p + k0 + 16);
        a1.u[0] = *(const uint4*)(a1p + k0);
        a1.u[1] = *(const uint4*)(a1p + k0 + 16);
        // B 32x16 fragment: same K order as A (lane>=16 -> +8)
        // (folded into b*p); B[k][n] = W[n][k] since we compute act @ W^T.
        b0.u[0] = *(const uint4*)(b0p + k0);
        b0.u[1] = *(const uint4*)(b0p + k0 + 16);
        b1.u[0] = *(const uint4*)(b1p + k0);
        b1.u[1] = *(const uint4*)(b1p + k0 + 16);
        if (k0 + 32 < K) {  // pull next weight chunk toward L0 (global_prefetch_b8)
            __builtin_prefetch(b0p + k0 + 32, 0, 1);
            __builtin_prefetch(b1p + k0 + 32, 0, 1);
        }
        acc[0][0] = __builtin_amdgcn_wmma_f32_16x16x32_f16(
            false, a0.v, false, b0.v, (short)0, acc[0][0], false, false);
        acc[0][1] = __builtin_amdgcn_wmma_f32_16x16x32_f16(
            false, a0.v, false, b1.v, (short)0, acc[0][1], false, false);
        acc[1][0] = __builtin_amdgcn_wmma_f32_16x16x32_f16(
            false, a1.v, false, b0.v, (short)0, acc[1][0], false, false);
        acc[1][1] = __builtin_amdgcn_wmma_f32_16x16x32_f16(
            false, a1.v, false, b1.v, (short)0, acc[1][1], false, false);
    }
}

// ---------------------------------------------------------------------------
// Fused LSTM pre-activation GEMM:  G = X @ Wih^T + H @ Whh^T
//   X : M x K1 (row stride ldx, f16)      Wih : N x K1 (row-major, f16)
//   H : M x K2 (row stride ldh, f16)      Whh : N x K2 (row-major, f16)
//   G : M x N  (row-major, f32)           M == 256; N,K1,K2 multiples of 32
// Grid: (N/256, M/32), 256 threads = 8 waves; wave w -> 32-col group
// blockIdx.x*8+w, blockIdx.y -> 32-row group.
// ---------------------------------------------------------------------------
__global__ __launch_bounds__(256)
void lstm_step_gemm(const half_t* __restrict__ X, int ldx,
                    const half_t* __restrict__ Wih, int K1,
                    const half_t* __restrict__ Hst, int ldh,
                    const half_t* __restrict__ Whh, int K2,
                    float* __restrict__ G, int N) {
    const int lane = threadIdx.x & 31;
    const int wave = threadIdx.x >> 5;
    const int tn   = blockIdx.x * 8 + wave;   // 32-wide n group
    const int tm   = blockIdx.y;              // 32-tall m group
    const int l15  = lane & 15;
    const int sel  = lane >> 4;               // lane half (0|1)
    const int m0   = tm * 32 + l15;
    const int m1   = m0 + 16;
    const int n0   = tn * 32 + l15;
    const int n1   = n0 + 16;

    v8f acc[2][2];
#pragma unroll
    for (int mi = 0; mi < 2; ++mi)
#pragma unroll
        for (int ni = 0; ni < 2; ++ni)
#pragma unroll
            for (int j = 0; j < 8; ++j) acc[mi][ni][j] = 0.0f;

    // X @ Wih^T contribution
    wmma2x2_k_loop(X   + (size_t)m0 * ldx + sel * 8,
                   X   + (size_t)m1 * ldx + sel * 8,
                   Wih + (size_t)n0 * K1  + sel * 8,
                   Wih + (size_t)n1 * K1  + sel * 8, K1, acc);
    // H @ Whh^T contribution
    wmma2x2_k_loop(Hst + (size_t)m0 * ldh + sel * 8,
                   Hst + (size_t)m1 * ldh + sel * 8,
                   Whh + (size_t)n0 * K2  + sel * 8,
                   Whh + (size_t)n1 * K2  + sel * 8, K2, acc);

    // D layout (ISA 7.12.2): VGPR j of tile (mi,ni) holds row
    // tm*32 + mi*16 + sel*8 + j, column tn*32 + ni*16 + l15.
#pragma unroll
    for (int mi = 0; mi < 2; ++mi) {
        const int mbase = tm * 32 + mi * 16 + sel * 8;
#pragma unroll
        for (int ni = 0; ni < 2; ++ni) {
            const int col = tn * 32 + ni * 16 + l15;
#pragma unroll
            for (int j = 0; j < 8; ++j)
                G[(size_t)(mbase + j) * N + col] = acc[mi][ni][j];
        }
    }
}

// ---------------------------------------------------------------------------
// Fused LSTM gates: g -> (h_new, c_new). Writes c (f32), h (f16 for the next
// WMMA step), optional sequence store (f16) and optional f32 output store.
// ---------------------------------------------------------------------------
__global__ __launch_bounds__(256)
void lstm_gates_kernel(const float* __restrict__ G,
                       const float* __restrict__ bih,
                       const float* __restrict__ bhh,
                       float* __restrict__ c,
                       half_t* __restrict__ h16,
                       int Hdim,
                       half_t* __restrict__ hs_store,
                       float* __restrict__ out_store, int out_stride) {
    int idx = blockIdx.x * blockDim.x + threadIdx.x;
    int total = B_ * Hdim;
    if (idx >= total) return;
    int b = idx / Hdim;
    int j = idx - b * Hdim;

    const float* g = G + (size_t)b * 4 * Hdim;
    float gi = g[j]            + bih[j]            + bhh[j];
    float gf = g[Hdim + j]     + bih[Hdim + j]     + bhh[Hdim + j];
    float gg = g[2 * Hdim + j] + bih[2 * Hdim + j] + bhh[2 * Hdim + j];
    float go = g[3 * Hdim + j] + bih[3 * Hdim + j] + bhh[3 * Hdim + j];

    float si = 1.0f / (1.0f + __expf(-gi));
    float sf = 1.0f / (1.0f + __expf(-gf));
    float so = 1.0f / (1.0f + __expf(-go));
    float cn = sf * c[idx] + si * tanhf(gg);
    float hn = so * tanhf(cn);

    c[idx]   = cn;
    h16[idx] = (half_t)hn;
    if (hs_store)  hs_store[idx] = (half_t)hn;
    if (out_store) out_store[(size_t)b * out_stride + j] = hn;
}

// ---------------------------------------------------------------------------
// Host orchestration
// ---------------------------------------------------------------------------
extern "C" void kernel_launch(void* const* d_in, const int* in_sizes, int n_in,
                              void* d_out, int out_size, void* d_ws, size_t ws_size,
                              hipStream_t stream) {
    (void)in_sizes; (void)n_in; (void)out_size; (void)ws_size;

    const float* x        = (const float*)d_in[0];
    const float* eW[2]    = { (const float*)d_in[1],  (const float*)d_in[5]  };
    const float* eU[2]    = { (const float*)d_in[2],  (const float*)d_in[6]  };
    const float* ebi[2]   = { (const float*)d_in[3],  (const float*)d_in[7]  };
    const float* ebh[2]   = { (const float*)d_in[4],  (const float*)d_in[8]  };
    const float* dW[2]    = { (const float*)d_in[9],  (const float*)d_in[13] };
    const float* dU[2]    = { (const float*)d_in[10], (const float*)d_in[14] };
    const float* dbi[2]   = { (const float*)d_in[11], (const float*)d_in[15] };
    const float* dbh[2]   = { (const float*)d_in[12], (const float*)d_in[16] };
    float* out = (float*)d_out;

    // ---- workspace carve-up (256B aligned) ----
    size_t off = 0;
    auto carve = [&](size_t bytes) -> void* {
        void* p = (char*)d_ws + off;
        off += (bytes + 255) & ~(size_t)255;
        return p;
    };
    half_t* xf   = (half_t*)carve((size_t)B_ * S_ * I_ * 2);        // x in f16
    half_t* hs0  = (half_t*)carve((size_t)S_ * B_ * H_ * 2);        // enc L0 seq out
    half_t* eW0f = (half_t*)carve((size_t)4 * H_ * I_ * 2);
    half_t* eU0f = (half_t*)carve((size_t)4 * H_ * H_ * 2);
    half_t* eW1f = (half_t*)carve((size_t)4 * I_ * H_ * 2);
    half_t* eU1f = (half_t*)carve((size_t)4 * I_ * I_ * 2);
    half_t* dW0f = (half_t*)carve((size_t)4 * H_ * I_ * 2);
    half_t* dU0f = (half_t*)carve((size_t)4 * H_ * H_ * 2);
    half_t* dW1f = (half_t*)carve((size_t)4 * I_ * H_ * 2);
    half_t* dU1f = (half_t*)carve((size_t)4 * I_ * I_ * 2);
    float*  g    = (float*) carve((size_t)B_ * 4 * H_ * 4);         // gate preact
    half_t* h0   = (half_t*)carve((size_t)B_ * H_ * 2);
    float*  c0   = (float*) carve((size_t)B_ * H_ * 4);
    half_t* h1   = (half_t*)carve((size_t)B_ * I_ * 2);
    float*  c1   = (float*) carve((size_t)B_ * I_ * 4);

    // ---- one-time conversions ----
    auto conv = [&](const float* s, half_t* d, int n) {
        f32_to_f16_kernel<<<(n + 255) / 256, 256, 0, stream>>>(s, d, n);
    };
    conv(x,     xf,   B_ * S_ * I_);
    conv(eW[0], eW0f, 4 * H_ * I_);
    conv(eU[0], eU0f, 4 * H_ * H_);
    conv(eW[1], eW1f, 4 * I_ * H_);
    conv(eU[1], eU1f, 4 * I_ * I_);
    conv(dW[0], dW0f, 4 * H_ * I_);
    conv(dU[0], dU0f, 4 * H_ * H_);
    conv(dW[1], dW1f, 4 * I_ * H_);
    conv(dU[1], dU1f, 4 * I_ * I_);

    // ---- zero initial states ----
    hipMemsetAsync(h0, 0, (size_t)B_ * H_ * 2, stream);
    hipMemsetAsync(c0, 0, (size_t)B_ * H_ * 4, stream);
    hipMemsetAsync(h1, 0, (size_t)B_ * I_ * 2, stream);
    hipMemsetAsync(c1, 0, (size_t)B_ * I_ * 4, stream);

    const int N0 = 4 * H_;            // 2048
    const int N1 = 4 * I_;            // 256
    const dim3 gemmBlk(256);
    const dim3 g0(N0 / 256, B_ / 32); // layer0 GEMM grid: (8, 8)
    const dim3 g1(N1 / 256, B_ / 32); // layer1 GEMM grid: (1, 8)
    const int gate0 = (B_ * H_ + 255) / 256;
    const int gate1 = (B_ * I_ + 255) / 256;

    // ---- encoder: interleave layer0/layer1 per timestep ----
    for (int t = 0; t < S_; ++t) {
        // layer 0: g = x_t @ Wih0^T + h0 @ Whh0^T
        lstm_step_gemm<<<g0, gemmBlk, 0, stream>>>(
            xf + (size_t)t * I_, S_ * I_, eW0f, I_,
            h0, H_, eU0f, H_, g, N0);
        lstm_gates_kernel<<<gate0, 256, 0, stream>>>(
            g, ebi[0], ebh[0], c0, h0, H_,
            hs0 + (size_t)t * B_ * H_, nullptr, 0);
        // layer 1: g = hs0_t @ Wih1^T + h1 @ Whh1^T
        lstm_step_gemm<<<g1, gemmBlk, 0, stream>>>(
            hs0 + (size_t)t * B_ * H_, H_, eW1f, H_,
            h1, I_, eU1f, I_, g, N1);
        lstm_gates_kernel<<<gate1, 256, 0, stream>>>(
            g, ebi[1], ebh[1], c1, h1, I_, nullptr, nullptr, 0);
    }
    // After the loop: h1 == x[:, -1] of the encoder stack (decoder's first
    // input), and (h0,c0),(h1,c1) are the decoder's initial hidden states.

    // ---- decoder: P autoregressive steps ----
    for (int t = 0; t < P_; ++t) {
        // layer 0: input is previous step's layer-1 output (h1)
        lstm_step_gemm<<<g0, gemmBlk, 0, stream>>>(
            h1, I_, dW0f, I_, h0, H_, dU0f, H_, g, N0);
        lstm_gates_kernel<<<gate0, 256, 0, stream>>>(
            g, dbi[0], dbh[0], c0, h0, H_, nullptr, nullptr, 0);
        // layer 1: input is this step's layer-0 output (h0)
        lstm_step_gemm<<<g1, gemmBlk, 0, stream>>>(
            h0, H_, dW1f, H_, h1, I_, dU1f, I_, g, N1);
        lstm_gates_kernel<<<gate1, 256, 0, stream>>>(
            g, dbi[1], dbh[1], c1, h1, I_, nullptr,
            out + (size_t)t * I_, P_ * I_);
    }
}
